// PolicyNetGARCH_33354716021605
// MI455X (gfx1250) — matrix operand order repaired and checked by reading, not compiled, not run on hardware
//
#include <hip/hip_runtime.h>
#include <hip/hip_bf16.h>

// ---------------------------------------------------------------------------
// MI455X (gfx1250) residual-LSTM policy net, wave-pipelined.
//   Recurrence-bound problem (34 GFLOP, ~70MB HBM): parallelism = batch tiles
//   x pipeline stages. 64 workgroups x 4 wave32 = 256 waves; wave k owns LSTM
//   block k, xt stream handed wave->wave via double-buffered LDS + barrier.
//   Gates via v_wmma_f32_16x16x32_f16 (K=32 == H), weights pre-swizzled into
//   LDS A-operand layout, C->B feedback via ds_swizzle SWAPX16, branchless
//   v_exp/v_rcp/v_rsq transcendentals (EXEC stays all-ones for WMMA).
// ---------------------------------------------------------------------------

typedef __attribute__((ext_vector_type(16))) _Float16 v16h;
typedef __attribute__((ext_vector_type(8)))  float    v8f;

constexpr int BB  = 1024;   // batch
constexpr int SS  = 512;    // sequence
constexpr int HH  = 32;     // hidden
constexpr int NB  = 4;      // lstm blocks == waves per workgroup
constexpr int FAN = 7;      // OBS + NINS

// LDS layout (bytes)
constexpr int AW_BYTES   = NB * 2 * 8 * 1024;       // 64 A-tiles, 32 lanes x 16 halfs
constexpr int BIAS_OFF   = AW_BYTES;                // [NB][128] f32
constexpr int RMSW_OFF   = BIAS_OFF + NB * 4 * HH * 4;
constexpr int XBUF_OFF   = RMSW_OFF + NB * HH * 4;  // [NB][2][512] f32 handoff slots
constexpr int SMEM_BYTES = XBUF_OFF + NB * 2 * 512 * 4;

__device__ __forceinline__ float swz16(float x) {
  // ds_swizzle_b32, group-of-32: xor=0x10, or=0, and=0x1f  (SWAPX16)
  return __int_as_float(__builtin_amdgcn_ds_swizzle(__float_as_int(x), 0x401F));
}
// branchless fast transcendentals on v_exp_f32 / v_rcp_f32
__device__ __forceinline__ float fexp(float x) {
  return __builtin_amdgcn_exp2f(1.4426950408889634f * x);
}
__device__ __forceinline__ float fsig(float x) {
  return __builtin_amdgcn_rcpf(1.0f + fexp(-x));
}
__device__ __forceinline__ float ftanh(float x) {
  // tanh(x) = 1 - 2/(e^{2x}+1); saturates correctly at +-1
  return 1.0f - 2.0f * __builtin_amdgcn_rcpf(fexp(2.0f * x) + 1.0f);
}

__device__ __forceinline__ v8f bias_tile(const float* bb, int t, int lh) {
  float4 a = *(const float4*)(bb + t * 16 + lh * 8);
  float4 c = *(const float4*)(bb + t * 16 + lh * 8 + 4);
  v8f r;
  r[0]=a.x; r[1]=a.y; r[2]=a.z; r[3]=a.w;
  r[4]=c.x; r[5]=c.y; r[6]=c.z; r[7]=c.w;
  return r;
}

// ---- prep: xw[s,b,h] = concat(obs, prev_actions) @ W_in^T + b_in ----------
__global__ void xproj_kernel(const float* __restrict__ obs,
                             const float* __restrict__ prev,
                             const float* __restrict__ W_in,
                             const float* __restrict__ b_in,
                             float* __restrict__ xw) {
  int tid = blockIdx.x * blockDim.x + threadIdx.x;
  int h   = tid & (HH - 1);
  int bs  = tid >> 5;
  int s   = bs & (SS - 1);
  int b   = bs >> 9;              // SS == 512
  if (b >= BB) return;
  const float* w  = W_in + h * FAN;
  const float* ob = obs  + ((size_t)b * SS + s) * 5;
  const float* pv = prev + ((size_t)b * SS + s) * 2;
  float acc = b_in[h];
  acc += ob[0]*w[0] + ob[1]*w[1] + ob[2]*w[2] + ob[3]*w[3] + ob[4]*w[4];
  acc += pv[0]*w[5] + pv[1]*w[6];
  xw[(size_t)s * BB * HH + (size_t)b * HH + h] = acc;
}

// ---- main: 4-wave pipeline over blocks, scan over S --------------------
__global__ __launch_bounds__(NB * 32, 1)
void lstm_scan_kernel(const float* __restrict__ xw,     // [S,B,H]
                      const float* __restrict__ W_ih,   // [NB,4H,H]
                      const float* __restrict__ W_hh,   // [NB,4H,H]
                      const float* __restrict__ b_ih,   // [NB,4H]
                      const float* __restrict__ b_hh,   // [NB,4H]
                      const float* __restrict__ rms_w,  // [NB,H]
                      const float* __restrict__ head_w, // [2,H]
                      const float* __restrict__ head_b, // [2]
                      float* __restrict__ out) {
  extern __shared__ char smem[];
  v16h*  aWv   = (v16h*)smem;                  // [NB][2][8] tiles x 32 lanes
  float* biasL = (float*)(smem + BIAS_OFF);    // [NB][128]
  float* rmswL = (float*)(smem + RMSW_OFF);    // [NB][32]
  float* xbuf  = (float*)(smem + XBUF_OFF);    // [NB][2][512]

  const int lane = threadIdx.x & 31;
  const int k    = threadIdx.x >> 5;           // pipeline stage == lstm block
  const int lh   = lane >> 4;                  // lane half
  const int col  = lane & 15;
  const int b    = blockIdx.x * 16 + col;      // this lane's batch column (N)

  // ---- wave k builds its own block's A-operand tiles in LDS ----
  // A 16x32 f16, contiguous k order used here (B matches): lane<16 -> K {0..15}; lane>=16 -> K {16..31}
  for (int mat = 0; mat < 2; ++mat) {
    const float* W = (mat == 0 ? W_ih : W_hh) + (size_t)k * 4 * HH * HH;
    for (int t = 0; t < 8; ++t) {
      int m = t * 16 + col;                    // gate row
      v16h tmp;
      #pragma unroll
      for (int e = 0; e < 16; ++e) {
        int kk = lh * 16 + e;
        tmp[e] = (_Float16)W[m * HH + kk];
      }
      aWv[(k * 16 + mat * 8 + t) * 32 + lane] = tmp;
    }
  }
  for (int g = threadIdx.x; g < NB * 4 * HH; g += NB * 32)
    biasL[g] = b_ih[g] + b_hh[g];
  for (int g = threadIdx.x; g < NB * HH; g += NB * 32)
    rmswL[g] = rms_w[g];
  __syncthreads();

  // per-lane constants held in registers
  float rw[16];
  #pragma unroll
  for (int e = 0; e < 16; ++e) rw[e] = rmswL[k * HH + lh * 16 + e];
  float hw0[16], hw1[16];
  #pragma unroll
  for (int e = 0; e < 16; ++e) {
    hw0[e] = head_w[lh * 16 + e];
    hw1[e] = head_w[HH + lh * 16 + e];
  }
  const float hb  = lh ? head_b[1] : head_b[0];
  const float lim = lh ? 10.0f : 5.0f;
  const float* bb = biasL + k * 4 * HH;

  // block-k state: c in C-tile layout (h = 16j + lh*8 + r), h in B layout
  float cst[16], hst[16];
  #pragma unroll
  for (int e = 0; e < 16; ++e) { cst[e] = 0.0f; hst[e] = 0.0f; }

  float* hT = out + (size_t)2 * BB * SS;
  float* cT = hT + (size_t)NB * BB * HH;

  for (int it = 0; it < SS + NB - 1; ++it) {
    const int s = it - k;
    if (s >= 0 && s < SS) {            // uniform per wave -> EXEC stays full
      // ---- fetch xt (B layout: k-dim = lh*16+e, batch col b) ----
      float xt[16];
      if (k == 0) {
        const float* xp = xw + ((size_t)s * BB + b) * HH + lh * 16;
        float4 x0 = *(const float4*)(xp + 0);
        float4 x1 = *(const float4*)(xp + 4);
        float4 x2 = *(const float4*)(xp + 8);
        float4 x3 = *(const float4*)(xp + 12);
        xt[0]=x0.x; xt[1]=x0.y; xt[2]=x0.z; xt[3]=x0.w;
        xt[4]=x1.x; xt[5]=x1.y; xt[6]=x1.z; xt[7]=x1.w;
        xt[8]=x2.x; xt[9]=x2.y; xt[10]=x2.z; xt[11]=x2.w;
        xt[12]=x3.x; xt[13]=x3.y; xt[14]=x3.z; xt[15]=x3.w;
        if (s + 1 < SS) __builtin_prefetch(xp + (size_t)BB * HH, 0, 0);
      } else {
        const float* xin = xbuf + ((k - 1) * 2 + ((it + 1) & 1)) * 512 + lane * 16;
        float4 x0 = *(const float4*)(xin + 0);
        float4 x1 = *(const float4*)(xin + 4);
        float4 x2 = *(const float4*)(xin + 8);
        float4 x3 = *(const float4*)(xin + 12);
        xt[0]=x0.x; xt[1]=x0.y; xt[2]=x0.z; xt[3]=x0.w;
        xt[4]=x1.x; xt[5]=x1.y; xt[6]=x1.z; xt[7]=x1.w;
        xt[8]=x2.x; xt[9]=x2.y; xt[10]=x2.z; xt[11]=x2.w;
        xt[12]=x3.x; xt[13]=x3.y; xt[14]=x3.z; xt[15]=x3.w;
      }

      // ---- RMS norm over H (partner-lane exchange completes the row) ----
      float ss = 0.0f;
      #pragma unroll
      for (int e = 0; e < 16; ++e) ss += xt[e] * xt[e];
      ss += swz16(ss);
      const float scale = __builtin_amdgcn_rsqf(ss * (1.0f / HH) + 1e-6f);

      v16h xb, hbv;
      #pragma unroll
      for (int e = 0; e < 16; ++e) {
        xb[e]  = (_Float16)(rw[e] * xt[e] * scale);
        hbv[e] = (_Float16)hst[e];
      }

      // ---- gates^T: two passes of 4 tiles (i,f,g,o pairing j, j+2, j+4, j+6)
      float hc0[8], hc1[8];
      #pragma unroll
      for (int j = 0; j < 2; ++j) {
        v8f Ci = bias_tile(bb, j,     lh);
        v8f Cf = bias_tile(bb, j + 2, lh);
        v8f Cg = bias_tile(bb, j + 4, lh);
        v8f Co = bias_tile(bb, j + 6, lh);
        const int base = k * 16;
        v16h a;
        a = aWv[(base + j)         * 32 + lane];
        Ci = __builtin_amdgcn_wmma_f32_16x16x32_f16(false, a, false, xb,  (short)0, Ci, false, false);
        a = aWv[(base + 8 + j)     * 32 + lane];
        Ci = __builtin_amdgcn_wmma_f32_16x16x32_f16(false, a, false, hbv, (short)0, Ci, false, false);
        a = aWv[(base + j + 2)     * 32 + lane];
        Cf = __builtin_amdgcn_wmma_f32_16x16x32_f16(false, a, false, xb,  (short)0, Cf, false, false);
        a = aWv[(base + 8 + j + 2) * 32 + lane];
        Cf = __builtin_amdgcn_wmma_f32_16x16x32_f16(false, a, false, hbv, (short)0, Cf, false, false);
        a = aWv[(base + j + 4)     * 32 + lane];
        Cg = __builtin_amdgcn_wmma_f32_16x16x32_f16(false, a, false, xb,  (short)0, Cg, false, false);
        a = aWv[(base + 8 + j + 4) * 32 + lane];
        Cg = __builtin_amdgcn_wmma_f32_16x16x32_f16(false, a, false, hbv, (short)0, Cg, false, false);
        a = aWv[(base + j + 6)     * 32 + lane];
        Co = __builtin_amdgcn_wmma_f32_16x16x32_f16(false, a, false, xb,  (short)0, Co, false, false);
        a = aWv[(base + 8 + j + 6) * 32 + lane];
        Co = __builtin_amdgcn_wmma_f32_16x16x32_f16(false, a, false, hbv, (short)0, Co, false, false);

        #pragma unroll
        for (int r = 0; r < 8; ++r) {
          float cn = fsig(Cf[r]) * cst[j * 8 + r] + fsig(Ci[r]) * ftanh(Cg[r]);
          cst[j * 8 + r] = cn;
          float hn = fsig(Co[r]) * ftanh(cn);
          if (j == 0) hc0[r] = hn; else hc1[r] = hn;
        }
      }

      // ---- C-layout -> B-layout h_new via partner-lane exchange ----
      #pragma unroll
      for (int r = 0; r < 8; ++r) {
        float send = lh ? hc0[r] : hc1[r];
        float recv = swz16(send);
        float lo = lh ? recv : hc0[r];   // k-dim = lh*16 + r
        float hi = lh ? hc1[r] : recv;   // k-dim = lh*16 + 8 + r
        hst[r]     = lo;
        hst[8 + r] = hi;
        xt[r]     += lo;                 // residual
        xt[8 + r] += hi;
      }

      if (k < NB - 1) {
        // ---- hand xt to the next pipeline stage ----
        float* xo = xbuf + (k * 2 + (it & 1)) * 512 + lane * 16;
        *(float4*)(xo + 0)  = make_float4(xt[0],  xt[1],  xt[2],  xt[3]);
        *(float4*)(xo + 4)  = make_float4(xt[4],  xt[5],  xt[6],  xt[7]);
        *(float4*)(xo + 8)  = make_float4(xt[8],  xt[9],  xt[10], xt[11]);
        *(float4*)(xo + 12) = make_float4(xt[12], xt[13], xt[14], xt[15]);
      } else {
        // ---- heads on final xt: dot over H, symexp, clip ----
        float d0 = 0.0f, d1 = 0.0f;
        #pragma unroll
        for (int e = 0; e < 16; ++e) { d0 += hw0[e] * xt[e]; d1 += hw1[e] * xt[e]; }
        d0 += swz16(d0);
        d1 += swz16(d1);
        float raw = (lh ? d1 : d0) + hb;
        float v = copysignf(fexp(fabsf(raw)) - 1.0f, raw);
        v = fminf(fmaxf(v, -lim), lim);
        out[(size_t)lh * BB * SS + (size_t)b * SS + s] = v;
      }
    }
    __syncthreads();   // double-buffer handoff fence (all waves participate)
  }

  // ---- final states for this wave's block ----
  float* hp = hT + ((size_t)k * BB + b) * HH + lh * 16;
  *(float4*)(hp + 0)  = make_float4(hst[0],  hst[1],  hst[2],  hst[3]);
  *(float4*)(hp + 4)  = make_float4(hst[4],  hst[5],  hst[6],  hst[7]);
  *(float4*)(hp + 8)  = make_float4(hst[8],  hst[9],  hst[10], hst[11]);
  *(float4*)(hp + 12) = make_float4(hst[12], hst[13], hst[14], hst[15]);
  float* cp = cT + ((size_t)k * BB + b) * HH + lh * 8;
  *(float4*)(cp + 0)  = make_float4(cst[0],  cst[1],  cst[2],  cst[3]);
  *(float4*)(cp + 4)  = make_float4(cst[4],  cst[5],  cst[6],  cst[7]);
  *(float4*)(cp + 16) = make_float4(cst[8],  cst[9],  cst[10], cst[11]);
  *(float4*)(cp + 20) = make_float4(cst[12], cst[13], cst[14], cst[15]);
}

extern "C" void kernel_launch(void* const* d_in, const int* in_sizes, int n_in,
                              void* d_out, int out_size, void* d_ws, size_t ws_size,
                              hipStream_t stream) {
  const float* obs    = (const float*)d_in[0];
  const float* prev   = (const float*)d_in[1];
  const float* W_in   = (const float*)d_in[2];
  const float* b_in   = (const float*)d_in[3];
  const float* rms_w  = (const float*)d_in[4];
  const float* W_ih   = (const float*)d_in[5];
  const float* W_hh   = (const float*)d_in[6];
  const float* b_ih   = (const float*)d_in[7];
  const float* b_hh   = (const float*)d_in[8];
  const float* head_w = (const float*)d_in[9];
  const float* head_b = (const float*)d_in[10];
  float* outp = (float*)d_out;
  float* xw   = (float*)d_ws;   // [S,B,H] f32 = 64 MB scratch

  int total = BB * SS * HH;
  xproj_kernel<<<total / 256, 256, 0, stream>>>(obs, prev, W_in, b_in, xw);

  lstm_scan_kernel<<<BB / 16, NB * 32, SMEM_BYTES, stream>>>(
      xw, W_ih, W_hh, b_ih, b_hh, rms_w, head_w, head_b, outp);
}
